// Encoder_88072599372129
// MI455X (gfx1250) — matrix operand order repaired and checked by reading, not compiled, not run on hardware
//
#include <hip/hip_runtime.h>

// ---------------------------------------------------------------------------
// Types
// ---------------------------------------------------------------------------
typedef __attribute__((ext_vector_type(16))) __bf16          v16bf;
typedef __attribute__((ext_vector_type(8)))  __bf16          v8bf;
typedef __attribute__((ext_vector_type(8)))  float           v8f;
typedef __attribute__((ext_vector_type(8)))  unsigned short  us8;
typedef __attribute__((ext_vector_type(4)))  unsigned short  us4;

static constexpr int Ecfg  = 1024;
static constexpr int FFcfg = 4096;
static constexpr int Bcfg  = 4;
static constexpr int Scfg  = 2048;
static constexpr int BScfg = Bcfg * Scfg;   // 8192 token rows

__device__ __forceinline__ unsigned short f2bf(float f) {
    unsigned int u = __float_as_uint(f);
    u += 0x7FFFu + ((u >> 16) & 1u);        // round-to-nearest-even
    return (unsigned short)(u >> 16);
}

// Async global -> LDS 16-byte copy (ASYNCcnt-tracked).  LDS byte address is the
// low 32 bits of the generic shared-memory pointer (flat->LDS truncation).
#define ASYNC_CP16(ldsPtr, gPtr)                                              \
    asm volatile("global_load_async_to_lds_b128 %0, %1, off"                  \
                 :: "v"((unsigned int)(unsigned long long)(ldsPtr)),          \
                    "v"((unsigned long long)(gPtr))                           \
                 : "memory")

#define WAIT_ASYNC0() asm volatile("s_wait_asynccnt 0x0" ::: "memory")

// ---------------------------------------------------------------------------
// f32 -> bf16 conversion (weights)
// ---------------------------------------------------------------------------
__global__ __launch_bounds__(256) void cvt_kernel(const float* __restrict__ in,
                                                  unsigned short* __restrict__ out,
                                                  int n) {
    int i = blockIdx.x * 256 + threadIdx.x;
    if (i < n) out[i] = f2bf(in[i]);
}

// ---------------------------------------------------------------------------
// LayerNorm over E=1024, one row per block (256 threads x 4 elems)
// ---------------------------------------------------------------------------
template <bool OUT_BF16>
__global__ __launch_bounds__(256) void ln_kernel(const float* __restrict__ x,
                                                 const float* __restrict__ g,
                                                 const float* __restrict__ b,
                                                 void* __restrict__ out) {
    const int row = blockIdx.x;
    const int t   = threadIdx.x;
    const float4 v = ((const float4*)(x + (size_t)row * Ecfg))[t];

    float s  = v.x + v.y + v.z + v.w;
    float sq = v.x * v.x + v.y * v.y + v.z * v.z + v.w * v.w;

    __shared__ float r1[256], r2[256];
    r1[t] = s; r2[t] = sq;
    __syncthreads();
    for (int o = 128; o > 0; o >>= 1) {
        if (t < o) { r1[t] += r1[t + o]; r2[t] += r2[t + o]; }
        __syncthreads();
    }
    const float mean = r1[0] * (1.0f / Ecfg);
    const float var  = r2[0] * (1.0f / Ecfg) - mean * mean;
    const float rstd = rsqrtf(var + 1e-5f);

    const float4 gg = ((const float4*)g)[t];
    const float4 bb = ((const float4*)b)[t];
    float y0 = (v.x - mean) * rstd * gg.x + bb.x;
    float y1 = (v.y - mean) * rstd * gg.y + bb.y;
    float y2 = (v.z - mean) * rstd * gg.z + bb.z;
    float y3 = (v.w - mean) * rstd * gg.w + bb.w;

    if constexpr (OUT_BF16) {
        us4 o4; o4.x = f2bf(y0); o4.y = f2bf(y1); o4.z = f2bf(y2); o4.w = f2bf(y3);
        ((us4*)out)[(size_t)row * 256 + t] = o4;
    } else {
        float4 o4; o4.x = y0; o4.y = y1; o4.z = y2; o4.w = y3;
        ((float4*)out)[(size_t)row * 256 + t] = o4;
    }
}

// ---------------------------------------------------------------------------
// Row softmax over S=2048, f32 in -> bf16 out (256 threads x 8 elems in regs)
// ---------------------------------------------------------------------------
__global__ __launch_bounds__(256) void softmax_kernel(const float* __restrict__ sc,
                                                      unsigned short* __restrict__ p) {
    const int row = blockIdx.x;
    const int t   = threadIdx.x;
    const float* r = sc + (size_t)row * Scfg;

    float v[8];
#pragma unroll
    for (int i = 0; i < 8; i++) v[i] = r[t + i * 256];

    float m = v[0];
#pragma unroll
    for (int i = 1; i < 8; i++) m = fmaxf(m, v[i]);

    __shared__ float red[256];
    red[t] = m;
    __syncthreads();
    for (int o = 128; o > 0; o >>= 1) {
        if (t < o) red[t] = fmaxf(red[t], red[t + o]);
        __syncthreads();
    }
    m = red[0];
    __syncthreads();

    float s = 0.0f;
#pragma unroll
    for (int i = 0; i < 8; i++) { v[i] = __expf(v[i] - m); s += v[i]; }
    red[t] = s;
    __syncthreads();
    for (int o = 128; o > 0; o >>= 1) {
        if (t < o) red[t] += red[t + o];
        __syncthreads();
    }
    const float inv = 1.0f / red[0];

    unsigned short* pr = p + (size_t)row * Scfg;
#pragma unroll
    for (int i = 0; i < 8; i++) pr[t + i * 256] = f2bf(v[i] * inv);
}

// ---------------------------------------------------------------------------
// WMMA bf16 GEMM:  C(M,N) = scale * A(M,K) * B + bias + res,  B[k][n] = W[n][k]
//   A : bf16 row-major (M,K);  W : bf16 row-major (N,K)
//   OMODE: 0 = f32 store, 1 = bf16 store, 2 = bf16 transposed store ((N,Sb)/batch)
//   Block = 256 threads (8 waves), tile 128x128, K-step 32.
//   A/B K-tiles double-buffered in LDS via global_load_async_to_lds_b128;
//   row stride padded to 40 elems (80 B) to spread LDS bank starts.
// ---------------------------------------------------------------------------
#define LDSTRIDE 40

template <int OMODE, int RELU>
__global__ __launch_bounds__(256)
void gemm_wmma_bf16(const unsigned short* __restrict__ A,
                    const unsigned short* __restrict__ W,
                    const float* __restrict__ bias,
                    const float* __restrict__ res,
                    void* __restrict__ outp,
                    int N, int K, float scale,
                    long long sA, long long sW, long long sC,
                    int transSb) {
    __shared__ __align__(16) unsigned short smemA[2][128 * LDSTRIDE];
    __shared__ __align__(16) unsigned short smemB[2][128 * LDSTRIDE];

    const int z = blockIdx.z;
    A += (size_t)z * sA;
    W += (size_t)z * sW;

    const int t    = threadIdx.x;
    const int lane = t & 31;
    const int wv   = t >> 5;
    const int m0   = blockIdx.y * 128;
    const int n0   = blockIdx.x * 128;
    const int wm   = (wv >> 2) * 64;     // wave M offset in tile (2 waves on M)
    const int wn   = (wv & 3) * 32;      // wave N offset in tile (4 waves on N)
    const int lr   = lane & 15;
    const int ah   = (lane >> 4) << 3;   // A K-half offset: 0 / 8
    const int bh   = (lane >> 4) << 3;   // B K-half offset: 0 / 8

    // Cooperative async copy: tile = 128 rows x 64 B = 512 chunks of 16 B,
    // 2 chunks/thread, coalesced on the global side.
    const int r0 = t >> 2;               // rows r0 and r0+64
    const int cp = t & 3;                // 16 B chunk within row
    const unsigned short* gA0 = A + (size_t)(m0 + r0) * K + cp * 8;
    const unsigned short* gA1 = A + (size_t)(m0 + r0 + 64) * K + cp * 8;
    const unsigned short* gB0 = W + (size_t)(n0 + r0) * K + cp * 8;
    const unsigned short* gB1 = W + (size_t)(n0 + r0 + 64) * K + cp * 8;
    const int l0 = r0 * LDSTRIDE + cp * 8;
    const int l1 = (r0 + 64) * LDSTRIDE + cp * 8;

    auto issue = [&](int buf, int k0) {
        ASYNC_CP16(&smemA[buf][l0], gA0 + k0);
        ASYNC_CP16(&smemA[buf][l1], gA1 + k0);
        ASYNC_CP16(&smemB[buf][l0], gB0 + k0);
        ASYNC_CP16(&smemB[buf][l1], gB1 + k0);
    };

    v8f acc[4][2];
#pragma unroll
    for (int mi = 0; mi < 4; mi++)
#pragma unroll
        for (int ni = 0; ni < 2; ni++)
#pragma unroll
            for (int e = 0; e < 8; e++) acc[mi][ni][e] = 0.0f;

    issue(0, 0);

    for (int k0 = 0, it = 0; k0 < K; k0 += 32, ++it) {
        const int cur = it & 1;
        WAIT_ASYNC0();
        __syncthreads();                       // whole K-tile resident in LDS
        if (k0 + 32 < K) issue(cur ^ 1, k0 + 32);  // overlap DMA with WMMA

        v16bf af[4], bfr[2];
#pragma unroll
        for (int mi = 0; mi < 4; mi++) {
            const unsigned short* pa = &smemA[cur][(wm + mi * 16 + lr) * LDSTRIDE + ah];
            union { v16bf v; us8 h[2]; } u;
            u.h[0] = *(const us8*)(pa);        // K[half .. half+8)
            u.h[1] = *(const us8*)(pa + 16);   // K[half+16 .. half+24)
            af[mi] = u.v;
        }
#pragma unroll
        for (int ni = 0; ni < 2; ni++) {
            const unsigned short* pb = &smemB[cur][(wn + ni * 16 + lr) * LDSTRIDE + bh];
            union { v16bf v; us8 h[2]; } u;
            u.h[0] = *(const us8*)(pb);        // K[half .. half+8)
            u.h[1] = *(const us8*)(pb + 16);   // K[half+16 .. half+24)
            bfr[ni] = u.v;
        }
#pragma unroll
        for (int mi = 0; mi < 4; mi++)
#pragma unroll
            for (int ni = 0; ni < 2; ni++)
                acc[mi][ni] = __builtin_amdgcn_wmma_f32_16x16x32_bf16(
                    false, af[mi], false, bfr[ni], (short)0, acc[mi][ni],
                    false, false);

        __syncthreads();                       // reads done before buffer reuse
    }

    // Epilogue: C VGPR v -> row = base + v + 8*(lane>=16), col = n + lane%16
#pragma unroll
    for (int mi = 0; mi < 4; mi++) {
        const int rbase = m0 + wm + mi * 16 + ((lane >> 4) << 3);
#pragma unroll
        for (int ni = 0; ni < 2; ni++) {
            const int col = n0 + wn + ni * 16 + lr;
            const float bv = bias ? bias[col] : 0.0f;
            float tv[8];
#pragma unroll
            for (int v = 0; v < 8; v++) {
                float x = acc[mi][ni][v] * scale + bv;
                if (RELU) x = fmaxf(x, 0.0f);
                if (res)  x += res[(size_t)(rbase + v) * N + col];
                tv[v] = x;
            }
            if (OMODE == 0) {
                float* Co = (float*)outp + (size_t)z * sC;
#pragma unroll
                for (int v = 0; v < 8; v++)
                    Co[(size_t)(rbase + v) * N + col] = tv[v];
            } else if (OMODE == 1) {
                unsigned short* Co = (unsigned short*)outp + (size_t)z * sC;
#pragma unroll
                for (int v = 0; v < 8; v++)
                    Co[(size_t)(rbase + v) * N + col] = f2bf(tv[v]);
            } else {
                // transposed bf16 store: out[batch][col][row%Sb], 8 rows contiguous
                unsigned short* Co = (unsigned short*)outp;
                const int bb = rbase / transSb;
                us8 pk;
#pragma unroll
                for (int v = 0; v < 8; v++) pk[v] = f2bf(tv[v]);
                *(us8*)(Co + (size_t)bb * N * transSb + (size_t)col * transSb +
                        (rbase - bb * transSb)) = pk;
            }
        }
    }
}

// ---------------------------------------------------------------------------
// Host-side orchestration
// ---------------------------------------------------------------------------
extern "C" void kernel_launch(void* const* d_in, const int* in_sizes, int n_in,
                              void* d_out, int out_size, void* d_ws, size_t ws_size,
                              hipStream_t stream) {
    const float* src  = (const float*)d_in[0];
    const float* Wq_w = (const float*)d_in[1];  const float* Wq_b = (const float*)d_in[2];
    const float* Wk_w = (const float*)d_in[3];  const float* Wk_b = (const float*)d_in[4];
    const float* Wv_w = (const float*)d_in[5];  const float* Wv_b = (const float*)d_in[6];
    const float* Wo_w = (const float*)d_in[7];  const float* Wo_b = (const float*)d_in[8];
    const float* W1_w = (const float*)d_in[9];  const float* W1_b = (const float*)d_in[10];
    const float* W2_w = (const float*)d_in[11]; const float* W2_b = (const float*)d_in[12];
    const float* g1 = (const float*)d_in[13];   const float* b1 = (const float*)d_in[14];
    const float* g2 = (const float*)d_in[15];   const float* b2 = (const float*)d_in[16];
    const float* g3 = (const float*)d_in[17];   const float* b3 = (const float*)d_in[18];

    const int E = Ecfg, FF = FFcfg, Bb = Bcfg, S = Scfg, BS = BScfg;

    char* ws = (char*)d_ws;
    size_t off = 0;
    auto alloc = [&](size_t bytes) -> void* {
        void* p = ws + off;
        off = (off + bytes + 255) & ~((size_t)255);
        return p;
    };

    unsigned short* xln = (unsigned short*)alloc((size_t)BS * E * 2);
    unsigned short* Wqb = (unsigned short*)alloc((size_t)E * E * 2);
    unsigned short* Wkb = (unsigned short*)alloc((size_t)E * E * 2);
    unsigned short* Wvb = (unsigned short*)alloc((size_t)E * E * 2);
    unsigned short* Wob = (unsigned short*)alloc((size_t)E * E * 2);
    unsigned short* W1b = (unsigned short*)alloc((size_t)FF * E * 2);
    unsigned short* W2b = (unsigned short*)alloc((size_t)E * FF * 2);
    unsigned short* Qb  = (unsigned short*)alloc((size_t)BS * E * 2);
    unsigned short* Kb  = (unsigned short*)alloc((size_t)BS * E * 2);
    unsigned short* Vt  = (unsigned short*)alloc((size_t)BS * E * 2);   // (B, E, S)
    float*          Sc  = (float*)alloc((size_t)Bb * S * S * 4);        // reused as F1 (bf16)
    unsigned short* P   = (unsigned short*)alloc((size_t)Bb * S * S * 2); // reused as OP (f32)
    float*          h   = (float*)alloc((size_t)BS * E * 4);

    unsigned short* AVb = Qb;                  // reuse: Q dead after scores
    unsigned short* hnb = Kb;                  // reuse: K dead after scores
    unsigned short* F1  = (unsigned short*)Sc; // reuse: scores dead after softmax
    float*          OP  = (float*)P;           // reuse: probs dead after AV GEMM

    // 1. x = LN(src) -> bf16
    ln_kernel<true><<<BS, 256, 0, stream>>>(src, g1, b1, xln);

    // 2. weights f32 -> bf16
    auto conv = [&](const float* s, unsigned short* d, int n) {
        cvt_kernel<<<(n + 255) / 256, 256, 0, stream>>>(s, d, n);
    };
    conv(Wq_w, Wqb, E * E);  conv(Wk_w, Wkb, E * E);
    conv(Wv_w, Wvb, E * E);  conv(Wo_w, Wob, E * E);
    conv(W1_w, W1b, FF * E); conv(W2_w, W2b, E * FF);

    // 3-5. Q, K (bf16 row-major), V (bf16 transposed per batch)
    gemm_wmma_bf16<1, 0><<<dim3(E / 128, BS / 128, 1), 256, 0, stream>>>(
        xln, Wqb, Wq_b, nullptr, Qb, E, E, 1.0f, 0, 0, 0, 0);
    gemm_wmma_bf16<1, 0><<<dim3(E / 128, BS / 128, 1), 256, 0, stream>>>(
        xln, Wkb, Wk_b, nullptr, Kb, E, E, 1.0f, 0, 0, 0, 0);
    gemm_wmma_bf16<2, 0><<<dim3(E / 128, BS / 128, 1), 256, 0, stream>>>(
        xln, Wvb, Wv_b, nullptr, Vt, E, E, 1.0f, 0, 0, 0, S);

    // 6. scores = (Q K^T) / sqrt(E)  per batch -> f32 (B,S,S)
    gemm_wmma_bf16<0, 0><<<dim3(S / 128, S / 128, Bb), 256, 0, stream>>>(
        Qb, Kb, nullptr, nullptr, Sc, S, E, 0.03125f,
        (long long)S * E, (long long)S * E, (long long)S * S, 0);

    // 7. softmax rows -> bf16 probs
    softmax_kernel<<<Bb * S, 256, 0, stream>>>(Sc, P);

    // 8. AV = probs @ V  per batch -> bf16 (B,S,E)   ("W" = V^T, rows E-major)
    gemm_wmma_bf16<1, 0><<<dim3(E / 128, S / 128, Bb), 256, 0, stream>>>(
        P, Vt, nullptr, nullptr, AVb, E, S, 1.0f,
        (long long)S * S, (long long)E * S, (long long)S * E, 0);

    // 9. h = AV @ Wo^T + bo + src -> f32
    gemm_wmma_bf16<0, 0><<<dim3(E / 128, BS / 128, 1), 256, 0, stream>>>(
        AVb, Wob, Wo_b, src, h, E, E, 1.0f, 0, 0, 0, 0);

    // 10. hn = LN(h) -> bf16
    ln_kernel<true><<<BS, 256, 0, stream>>>(h, g2, b2, hnb);

    // 11. F1 = relu(hn @ W1^T + b1) -> bf16 (B,S,FF)
    gemm_wmma_bf16<1, 1><<<dim3(FF / 128, BS / 128, 1), 256, 0, stream>>>(
        hnb, W1b, W1_b, nullptr, F1, FF, E, 1.0f, 0, 0, 0, 0);

    // 12. OP = F1 @ W2^T + b2 + h -> f32
    gemm_wmma_bf16<0, 0><<<dim3(E / 128, BS / 128, 1), 256, 0, stream>>>(
        F1, W2b, W2_b, h, OP, E, FF, 1.0f, 0, 0, 0, 0);

    // 13. out = LN(OP) -> f32 d_out
    ln_kernel<false><<<BS, 256, 0, stream>>>(OP, g3, b3, d_out);

    (void)in_sizes; (void)n_in; (void)out_size; (void)ws_size;
}
